// TransformerBlock_51290499449330
// MI455X (gfx1250) — matrix operand order repaired and checked by reading, not compiled, not run on hardware
//
#include <hip/hip_runtime.h>
#include <hip/hip_bf16.h>

// ---------------------------------------------------------------------------
// CDNA5 (gfx1250) transformer block: bf16 WMMA GEMMs + flash attention,
// TDM (tensor_load_to_lds) pipelined tile staging.
// wave32, 8 waves / 256-thread workgroups.
// ---------------------------------------------------------------------------

typedef __attribute__((ext_vector_type(16))) __bf16 v16bf;
typedef __attribute__((ext_vector_type(8)))  float  v8f;
typedef unsigned int u32x4 __attribute__((ext_vector_type(4)));
typedef int          i32x4 __attribute__((ext_vector_type(4)));
typedef int          i32x8 __attribute__((ext_vector_type(8)));

#if defined(__HIP_DEVICE_COMPILE__) && __has_builtin(__builtin_amdgcn_tensor_load_to_lds)
#define HAVE_TDM 1
#else
#define HAVE_TDM 0
#endif

__device__ __forceinline__ unsigned short f2bf(float f) {
  union { float f; unsigned int u; } c; c.f = f;
  unsigned int u = c.u;
  return (unsigned short)((u + 0x7FFFu + ((u >> 16) & 1u)) >> 16);
}

__device__ __forceinline__ v8f wmma_bf16(v16bf a, v16bf b, v8f c) {
  return __builtin_amdgcn_wmma_f32_16x16x32_bf16(false, a, false, b, (short)0, c,
                                                 false, false);
}

// A fragment (16x32 bf16 slice): lane row = rc, K bytes: kcByte+16*half and +32.
__device__ __forceinline__ v16bf load_a_frag(const unsigned short* base, int rowBytes,
                                             int row, int kcByte, int half) {
  union { v16bf v; uint4 q[2]; } f;
  const char* p = (const char*)base + (size_t)row * rowBytes + kcByte + 16 * half;
  f.q[0] = *(const uint4*)p;
  f.q[1] = *(const uint4*)(p + 32);
  return f.v;
}

// B fragment from K-transposed LDS tile Bs[n][k]: lane col = rc.
__device__ __forceinline__ v16bf load_b_frag(const unsigned short* base, int rowBytes,
                                             int col, int kcByte, int half) {
  union { v16bf v; uint4 q[2]; } f;
  const char* p = (const char*)base + (size_t)col * rowBytes + kcByte + 16 * half;
  f.q[0] = *(const uint4*)p;
  f.q[1] = *(const uint4*)(p + 32);
  return f.v;
}

#if HAVE_TDM
// 2-D TDM descriptor: invariant words built once, per-issue patches only
// lds_addr + global_addr (3 dwords).
struct Tdm2D {
  u32x4 g0; i32x8 g1; i32x4 g2; i32x4 g3;
#if __clang_major__ >= 23
  i32x8 g4;
#endif
  __device__ __forceinline__ void init(unsigned tile_d0, unsigned tile_d1,
                                       unsigned long long stride0) {
    u32x4 a = {1u, 0u, 0u, 2u << 30};            // count=1 | type=2
    g0 = a;
    i32x8 b1;
    b1[0] = 0x00010000;                          // data_size = 2B
    b1[1] = (int)((tile_d0 & 0xFFFFu) << 16);    // tensor_dim0 lo16
    b1[2] = (int)((tile_d0 >> 16) | ((tile_d1 & 0xFFFFu) << 16));
    b1[3] = (int)((tile_d1 >> 16) | (tile_d0 << 16));   // | tile_dim0
    b1[4] = (int)(tile_d1 & 0xFFFFu);            // tile_dim1 (tile_dim2=0)
    b1[5] = (int)(stride0 & 0xFFFFFFFFu);
    b1[6] = (int)((stride0 >> 32) & 0xFFFFu);
    b1[7] = 0;
    g1 = b1;
    i32x4 z = {0, 0, 0, 0};
    g2 = z; g3 = z;
#if __clang_major__ >= 23
    i32x8 z8 = {0, 0, 0, 0, 0, 0, 0, 0};
    g4 = z8;
#endif
  }
  __device__ __forceinline__ void issue(unsigned lds_off, const void* gaddr) const {
    unsigned long long ga = (unsigned long long)(uintptr_t)gaddr;
    u32x4 h0 = g0;
    h0[1] = lds_off;
    h0[2] = (unsigned)(ga & 0xFFFFFFFFu);
    h0[3] = (unsigned)((ga >> 32) & 0x01FFFFFFu) | (2u << 30);
#if __clang_major__ >= 23
    __builtin_amdgcn_tensor_load_to_lds(h0, g1, g2, g3, g4, 0);
#else
    __builtin_amdgcn_tensor_load_to_lds(h0, g1, g2, g3, 0);
#endif
  }
};
#endif

// ---------------------------------------------------------------------------
// Weight transpose + cast: W[K,N] fp32 -> Wt[N,K] bf16 (32x32 tiles)
// ---------------------------------------------------------------------------
__global__ __launch_bounds__(256) void transpose_cast_bf16(
    const float* __restrict__ W, unsigned short* __restrict__ Wt, int K, int N) {
  __shared__ unsigned short tile[32 * 33];
  const int t = threadIdx.x;
  const int n0 = blockIdx.x * 32, k0 = blockIdx.y * 32;
  const int c = t & 31, r = t >> 5;
#pragma unroll
  for (int i = 0; i < 4; ++i) {
    const int k = r + i * 8;
    tile[k * 33 + c] = f2bf(W[(size_t)(k0 + k) * N + n0 + c]);
  }
  __syncthreads();
#pragma unroll
  for (int i = 0; i < 4; ++i) {
    const int n = r + i * 8;
    Wt[(size_t)(n0 + n) * K + k0 + c] = tile[c * 33 + n];
  }
}

// ---------------------------------------------------------------------------
// V transpose: qkv[tok, 2048+h*64+d] bf16 -> vT[(b*16+h)*64+d, n] bf16
// ---------------------------------------------------------------------------
__global__ __launch_bounds__(256) void transpose_v_bf16(
    const unsigned short* __restrict__ qkv, unsigned short* __restrict__ vT) {
  __shared__ unsigned short tile[64 * 65];
  const int t = threadIdx.x;
  const int nb = blockIdx.x * 64;
  const int bh = blockIdx.y, b = bh >> 4, h = bh & 15;
  {
    const int row = t >> 2, c0 = (t & 3) * 16;
    const size_t src = ((size_t)b * 2048 + nb + row) * 3072 + 2048 + h * 64 + c0;
    union { uint4 q; unsigned short u[8]; } v0, v1;
    v0.q = *(const uint4*)&qkv[src];
    v1.q = *(const uint4*)&qkv[src + 8];
#pragma unroll
    for (int j = 0; j < 8; ++j) {
      tile[row * 65 + c0 + j] = v0.u[j];
      tile[row * 65 + c0 + 8 + j] = v1.u[j];
    }
  }
  __syncthreads();
  {
    const int d = t >> 2, n0 = (t & 3) * 16;
    union { uint4 q[2]; unsigned short u[16]; } o;
#pragma unroll
    for (int j = 0; j < 16; ++j) o.u[j] = tile[(n0 + j) * 65 + d];
    const size_t dst = ((size_t)bh * 64 + d) * 2048 + nb + n0;
    *(uint4*)&vT[dst]     = o.q[0];
    *(uint4*)&vT[dst + 8] = o.q[1];
  }
}

// ---------------------------------------------------------------------------
// LayerNorm over E=1024, one row per 256-thread block, bf16 output
// ---------------------------------------------------------------------------
__global__ __launch_bounds__(256) void layernorm_bf16(
    const float* __restrict__ x, const float* __restrict__ g,
    const float* __restrict__ bta, unsigned short* __restrict__ out, int E) {
  __shared__ float s1[256];
  __shared__ float s2[256];
  const int row = blockIdx.x;
  const int t = threadIdx.x;
  const float* xr = x + (size_t)row * E;
  float4 v = *(const float4*)&xr[t * 4];
  s1[t] = v.x + v.y + v.z + v.w;
  s2[t] = v.x * v.x + v.y * v.y + v.z * v.z + v.w * v.w;
  __syncthreads();
  for (int st = 128; st > 0; st >>= 1) {
    if (t < st) { s1[t] += s1[t + st]; s2[t] += s2[t + st]; }
    __syncthreads();
  }
  const float mu = s1[0] / E;
  const float var = s2[0] / E - mu * mu;
  const float rstd = rsqrtf(var + 1e-5f);
  const float4 gv = *(const float4*)&g[t * 4];
  const float4 bv = *(const float4*)&bta[t * 4];
  union { uint2 q; unsigned short u[4]; } o;
  o.u[0] = f2bf((v.x - mu) * rstd * gv.x + bv.x);
  o.u[1] = f2bf((v.y - mu) * rstd * gv.y + bv.y);
  o.u[2] = f2bf((v.z - mu) * rstd * gv.z + bv.z);
  o.u[3] = f2bf((v.w - mu) * rstd * gv.w + bv.w);
  *(uint2*)&out[(size_t)row * E + t * 4] = o.q;
}

// ---------------------------------------------------------------------------
// WMMA GEMM: C[M,N] = act(A[M,K] @ Bt[N,K]^T + bias) (+ resid)
// Tile BM=128, BN=128, BK=64; wave computes 32x64 (8 accums, 16 WMMA/stage).
// TDM double-buffered staging; branch-free steady-state pipeline.
// ---------------------------------------------------------------------------
__global__ __launch_bounds__(256) void gemm_bf16_wmma(
    const unsigned short* __restrict__ A, const unsigned short* __restrict__ Bt,
    const float* __restrict__ bias, const float* __restrict__ resid,
    float* __restrict__ outF, unsigned short* __restrict__ outH,
    int M, int N, int K, int act) {
  __shared__ unsigned short As[2][128 * 64];   // [m][k] per stage (16KB each)
  __shared__ unsigned short Bs[2][128 * 64];   // [n][k] per stage (16KB each)

  const int t = threadIdx.x;
  const int lane = t & 31, wave = t >> 5;
  const int half = lane >> 4, rc = lane & 15;
  const int wm = wave & 3;        // 32-row group
  const int wn = wave >> 2;       // 64-col group
  const int bm = blockIdx.y * 128;
  const int bn = blockIdx.x * 128;
  const int nk = K / 64;

  v8f zero = {0.f, 0.f, 0.f, 0.f, 0.f, 0.f, 0.f, 0.f};
  v8f acc0[4] = {zero, zero, zero, zero};
  v8f acc1[4] = {zero, zero, zero, zero};

#if HAVE_TDM
  Tdm2D td;
  const unsigned asLds[2] = {(unsigned)(uintptr_t)&As[0][0],
                             (unsigned)(uintptr_t)&As[1][0]};
  const unsigned bsLds[2] = {(unsigned)(uintptr_t)&Bs[0][0],
                             (unsigned)(uintptr_t)&Bs[1][0]};
  if (t < 32) {
    td.init(64, 128, (unsigned long long)K);
    td.issue(asLds[0], A + (size_t)bm * K);
    td.issue(bsLds[0], Bt + (size_t)bn * K);
  }
#endif

#define GEMM_COMPUTE(BUF)                                                     \
  _Pragma("unroll") for (int kc = 0; kc < 2; ++kc) {                          \
    v16bf a0 = load_a_frag(As[BUF], 128, wm * 32 + rc, kc * 64, half);        \
    v16bf a1 = load_a_frag(As[BUF], 128, wm * 32 + 16 + rc, kc * 64, half);   \
    _Pragma("unroll") for (int s = 0; s < 4; ++s) {                           \
      v16bf bb = load_b_frag(Bs[BUF], 128, wn * 64 + s * 16 + rc, kc * 64, half); \
      acc0[s] = wmma_bf16(a0, bb, acc0[s]);                                   \
      acc1[s] = wmma_bf16(a1, bb, acc1[s]);                                   \
    }                                                                         \
  }

#if HAVE_TDM
  // Steady state: always issue next stage, wait for current (2 outstanding).
  for (int kt = 0; kt < nk - 1; ++kt) {
    const int buf = kt & 1;
    if (t < 32) {
      const size_t ko = (size_t)(kt + 1) * 64;
      td.issue(asLds[buf ^ 1], A + (size_t)bm * K + ko);
      td.issue(bsLds[buf ^ 1], Bt + (size_t)bn * K + ko);
      __builtin_amdgcn_s_wait_tensorcnt(2);
    }
    __syncthreads();
    GEMM_COMPUTE(buf)
    __syncthreads();
  }
  {
    if (t < 32) __builtin_amdgcn_s_wait_tensorcnt(0);
    __syncthreads();
    const int buf = (nk - 1) & 1;
    GEMM_COMPUTE(buf)
  }
#else
  for (int kt = 0; kt < nk; ++kt) {
    const int buf = kt & 1;
    const int r2 = t >> 1, c2 = (t & 1) * 32;
#pragma unroll
    for (int j = 0; j < 4; ++j)
      *(uint4*)&As[buf][r2 * 64 + c2 + j * 8] =
          *(const uint4*)&A[(size_t)(bm + r2) * K + kt * 64 + c2 + j * 8];
#pragma unroll
    for (int j = 0; j < 4; ++j)
      *(uint4*)&Bs[buf][r2 * 64 + c2 + j * 8] =
          *(const uint4*)&Bt[(size_t)(bn + r2) * K + kt * 64 + c2 + j * 8];
    __syncthreads();
    GEMM_COMPUTE(buf)
    __syncthreads();
  }
#endif
#undef GEMM_COMPUTE

  // Epilogue: subtile row = 8*half + i, col = rc.
#pragma unroll
  for (int ms = 0; ms < 2; ++ms) {
#pragma unroll
    for (int s = 0; s < 4; ++s) {
      v8f acc = ms ? acc1[s] : acc0[s];
      const int rbase = bm + wm * 32 + ms * 16 + 8 * half;
      const int col = bn + wn * 64 + s * 16 + rc;
      const float bv = bias ? bias[col] : 0.f;
#pragma unroll
      for (int i = 0; i < 8; ++i) {
        float v = acc[i] + bv;
        if (act == 1) v = 0.5f * v * (1.f + erff(v * 0.70710678118654752f));
        const size_t idx = (size_t)(rbase + i) * N + col;
        if (resid) v += resid[idx];
        if (outF) outF[idx] = v;
        if (outH) outH[idx] = f2bf(v);
      }
    }
  }
}

// ---------------------------------------------------------------------------
// Flash attention (bf16 WMMA, fp32 online softmax, TDM-staged K/V tiles).
// qkv: [M,3072] bf16; vT: [(b*16+h)*64+d, 2048] bf16. Output bf16 [M,1024].
// grid: (32, 32); block 256.
// ---------------------------------------------------------------------------
__global__ __launch_bounds__(256) void flash_attn_wmma(
    const unsigned short* __restrict__ qkv, const unsigned short* __restrict__ vT,
    unsigned short* __restrict__ attn) {
  __shared__ unsigned short Qs[64 * 64];
  __shared__ unsigned short Ks[64 * 64];
  __shared__ unsigned short Vt[64 * 64];
  __shared__ unsigned short Ps[64 * 64];
  __shared__ float Sf[64 * 64];
  __shared__ float rowmax[64], rowsum[64], rowscale[64];

  const int t = threadIdx.x;
  const int lane = t & 31, wave = t >> 5;
  const int half = lane >> 4, rc = lane & 15;
  const int wm = wave & 3, wn = wave >> 2;
  const int qb = blockIdx.x;
  const int bh = blockIdx.y, b = bh >> 4, h = bh & 15;
  const size_t tok0 = (size_t)b * 2048 + (size_t)qb * 64;
  const size_t kbase = (size_t)b * 2048 * 3072 + 1024 + h * 64;   // K start, head h
  const size_t vbase = (size_t)bh * 64 * 2048;                    // vT start, head h

#if HAVE_TDM
  Tdm2D tdK, tdV;
  if (t < 32) {
    tdK.init(64, 64, 3072ull);
    tdV.init(64, 64, 2048ull);
    tdK.issue((unsigned)(uintptr_t)Ks, qkv + kbase);
    tdV.issue((unsigned)(uintptr_t)Vt, vT + vbase);
  }
#endif
  {
    const int row = t >> 2, d0 = (t & 3) * 16;
    const size_t src = (tok0 + row) * 3072 + h * 64 + d0;
    *(uint4*)&Qs[row * 64 + d0]     = *(const uint4*)&qkv[src];
    *(uint4*)&Qs[row * 64 + d0 + 8] = *(const uint4*)&qkv[src + 8];
  }
  if (t < 64) { rowmax[t] = -3.0e38f; rowsum[t] = 0.f; }

  v8f o0 = {0.f, 0.f, 0.f, 0.f, 0.f, 0.f, 0.f, 0.f};
  v8f o1 = o0;

  for (int kt = 0; kt < 32; ++kt) {
#if HAVE_TDM
    if (t < 32) __builtin_amdgcn_s_wait_tensorcnt(0);
    __syncthreads();
#else
    {
      const int row = t >> 2, d0 = (t & 3) * 16;
      const size_t src = ((size_t)b * 2048 + kt * 64 + row) * 3072 + 1024 + h * 64 + d0;
      *(uint4*)&Ks[row * 64 + d0]     = *(const uint4*)&qkv[src];
      *(uint4*)&Ks[row * 64 + d0 + 8] = *(const uint4*)&qkv[src + 8];
      const size_t vsrc = vbase + (size_t)row * 2048 + (size_t)kt * 64 + d0;
      *(uint4*)&Vt[row * 64 + d0]     = *(const uint4*)&vT[vsrc];
      *(uint4*)&Vt[row * 64 + d0 + 8] = *(const uint4*)&vT[vsrc + 8];
    }
    __syncthreads();
#endif

    // S = (Q @ K^T) * 1/sqrt(64)
    v8f s0 = {0.f, 0.f, 0.f, 0.f, 0.f, 0.f, 0.f, 0.f};
    v8f s1 = s0;
#pragma unroll
    for (int kc = 0; kc < 2; ++kc) {
      v16bf qf = load_a_frag(Qs, 128, wm * 16 + rc, kc * 64, half);
      v16bf k0 = load_b_frag(Ks, 128, wn * 32 + rc, kc * 64, half);
      v16bf k1 = load_b_frag(Ks, 128, wn * 32 + 16 + rc, kc * 64, half);
      s0 = wmma_bf16(qf, k0, s0);
      s1 = wmma_bf16(qf, k1, s1);
    }
#pragma unroll
    for (int i = 0; i < 8; ++i) {
      const int m = wm * 16 + 8 * half + i;
      Sf[m * 64 + wn * 32 + rc]      = s0[i] * 0.125f;
      Sf[m * 64 + wn * 32 + 16 + rc] = s1[i] * 0.125f;
    }
    __syncthreads();   // Sf complete; Ks free

#if HAVE_TDM
    if (t < 32 && kt + 1 < 32)
      tdK.issue((unsigned)(uintptr_t)Ks, qkv + kbase + (size_t)(kt + 1) * 64 * 3072);
#endif

    // Online softmax: 4 threads per row, shfl reductions.
    {
      const int row = t >> 2, q4 = t & 3;
      const float mx = rowmax[row];
      float tm = -3.0e38f;
#pragma unroll
      for (int j = 0; j < 16; ++j) tm = fmaxf(tm, Sf[row * 64 + q4 * 16 + j]);
      tm = fmaxf(tm, __shfl_xor(tm, 1, 32));
      tm = fmaxf(tm, __shfl_xor(tm, 2, 32));
      const float nm = fmaxf(mx, tm);
      float sum = 0.f;
#pragma unroll
      for (int j = 0; j < 16; ++j) {
        const float p = __expf(Sf[row * 64 + q4 * 16 + j] - nm);
        sum += p;
        Ps[row * 64 + q4 * 16 + j] = f2bf(p);
      }
      sum += __shfl_xor(sum, 1, 32);
      sum += __shfl_xor(sum, 2, 32);
      __builtin_amdgcn_wave_barrier();
      if (q4 == 0) {
        const float sc = __expf(mx - nm);
        rowsum[row] = rowsum[row] * sc + sum;
        rowmax[row] = nm;
        rowscale[row] = sc;
      }
    }
    __syncthreads();   // Ps + row state ready

    // O = O*scale + P @ V
#pragma unroll
    for (int i = 0; i < 8; ++i) {
      const float sc = rowscale[wm * 16 + 8 * half + i];
      o0[i] *= sc;
      o1[i] *= sc;
    }
#pragma unroll
    for (int kc = 0; kc < 2; ++kc) {
      v16bf pf = load_a_frag(Ps, 128, wm * 16 + rc, kc * 64, half);
      v16bf v0 = load_b_frag(Vt, 128, wn * 32 + rc, kc * 64, half);
      v16bf v1 = load_b_frag(Vt, 128, wn * 32 + 16 + rc, kc * 64, half);
      o0 = wmma_bf16(pf, v0, o0);
      o1 = wmma_bf16(pf, v1, o1);
    }
    __syncthreads();   // Vt, Ps free

#if HAVE_TDM
    if (t < 32 && kt + 1 < 32)
      tdV.issue((unsigned)(uintptr_t)Vt, vT + vbase + (size_t)(kt + 1) * 64);
#endif
  }

#pragma unroll
  for (int i = 0; i < 8; ++i) {
    const int m = wm * 16 + 8 * half + i;
    const float inv = 1.f / rowsum[m];
    const int d0c = wn * 32 + rc;
    attn[(tok0 + m) * 1024 + h * 64 + d0c]      = f2bf(o0[i] * inv);
    attn[(tok0 + m) * 1024 + h * 64 + d0c + 16] = f2bf(o1[i] * inv);
  }
}

// ---------------------------------------------------------------------------
// Host orchestration
// ---------------------------------------------------------------------------
extern "C" void kernel_launch(void* const* d_in, const int* in_sizes, int n_in,
                              void* d_out, int out_size, void* d_ws, size_t ws_size,
                              hipStream_t stream) {
  (void)in_sizes; (void)n_in; (void)out_size; (void)ws_size;
  const float* x      = (const float*)d_in[0];
  const float* ln1_g  = (const float*)d_in[1];
  const float* ln1_b  = (const float*)d_in[2];
  const float* qkv_w  = (const float*)d_in[3];
  const float* qkv_b  = (const float*)d_in[4];
  const float* proj_w = (const float*)d_in[5];
  const float* proj_b = (const float*)d_in[6];
  const float* ln2_g  = (const float*)d_in[7];
  const float* ln2_b  = (const float*)d_in[8];
  const float* fc1_w  = (const float*)d_in[9];
  const float* fc1_b  = (const float*)d_in[10];
  const float* fc2_w  = (const float*)d_in[11];
  const float* fc2_b  = (const float*)d_in[12];
  float* out = (float*)d_out;

  const int M = 4096, E = 1024, FF = 4096, E3 = 3072;

  char* ws = (char*)d_ws;
  unsigned short* wqT  = (unsigned short*)ws; ws += (size_t)E3 * E * 2;  // [3072,1024]
  unsigned short* wpT  = (unsigned short*)ws; ws += (size_t)E * E * 2;   // [1024,1024]
  unsigned short* w1T  = (unsigned short*)ws; ws += (size_t)FF * E * 2;  // [4096,1024]
  unsigned short* w2T  = (unsigned short*)ws; ws += (size_t)E * FF * 2;  // [1024,4096]
  unsigned short* hbuf = (unsigned short*)ws; ws += (size_t)M * E * 2;
  unsigned short* qkv  = (unsigned short*)ws; ws += (size_t)M * E3 * 2;
  unsigned short* vT   = (unsigned short*)ws; ws += (size_t)M * E * 2;
  unsigned short* attn = (unsigned short*)ws; ws += (size_t)M * E * 2;
  float*          x1   = (float*)ws;          ws += (size_t)M * E * 4;
  unsigned short* a1   = (unsigned short*)ws;

  // 1) Transpose + cast weights to [N,K] bf16
  transpose_cast_bf16<<<dim3(E3 / 32, E / 32), 256, 0, stream>>>(qkv_w, wqT, E, E3);
  transpose_cast_bf16<<<dim3(E / 32, E / 32), 256, 0, stream>>>(proj_w, wpT, E, E);
  transpose_cast_bf16<<<dim3(FF / 32, E / 32), 256, 0, stream>>>(fc1_w, w1T, E, FF);
  transpose_cast_bf16<<<dim3(E / 32, FF / 32), 256, 0, stream>>>(fc2_w, w2T, FF, E);

  // 2) LN1
  layernorm_bf16<<<M, 256, 0, stream>>>(x, ln1_g, ln1_b, hbuf, E);

  // 3) QKV GEMM -> bf16
  gemm_bf16_wmma<<<dim3(E3 / 128, M / 128), 256, 0, stream>>>(
      hbuf, wqT, qkv_b, nullptr, nullptr, qkv, M, E3, E, 0);

  // 4) V transpose, then flash attention -> bf16 [M,E]
  transpose_v_bf16<<<dim3(32, 32), 256, 0, stream>>>(qkv, vT);
  flash_attn_wmma<<<dim3(32, 32), 256, 0, stream>>>(qkv, vT, attn);

  // 5) Projection GEMM + residual(x) -> fp32 x1
  gemm_bf16_wmma<<<dim3(E / 128, M / 128), 256, 0, stream>>>(
      attn, wpT, proj_b, x, x1, nullptr, M, E, E, 0);

  // 6) LN2
  layernorm_bf16<<<M, 256, 0, stream>>>(x1, ln2_g, ln2_b, hbuf, E);

  // 7) FC1 + exact GELU -> bf16 a1
  gemm_bf16_wmma<<<dim3(FF / 128, M / 128), 256, 0, stream>>>(
      hbuf, w1T, fc1_b, nullptr, nullptr, a1, M, FF, E, 1);

  // 8) FC2 + residual(x1) -> fp32 out
  gemm_bf16_wmma<<<dim3(E / 128, M / 128), 256, 0, stream>>>(
      a1, w2T, fc2_b, x1, out, nullptr, M, E, FF, 0);
}
